// BatchTaskAlignedAssigner_86517821212361
// MI455X (gfx1250) — compile-verified, matrix operand order checked by reading, not run on hardware
//
#include <hip/hip_runtime.h>

#define NB   16
#define NG   64
#define NP   8400
#define NC   80
#define KTOP 13
#define TILE 256
#define NPT  33        /* ceil(NP/256) */
#define BGP  ((size_t)NB*NG*NP)
#define BP   (NB*NP)

typedef __attribute__((ext_vector_type(16))) _Float16 v16h;
typedef __attribute__((ext_vector_type(8)))  float    v8f;

// ---- CDNA5 async global->LDS copy (ASYNCcnt path) ------------------------
__device__ __forceinline__ void async_copy_b128(unsigned lds_off, const void* gaddr) {
  asm volatile("global_load_async_to_lds_b128 %0, %1, off"
               :: "v"(lds_off), "v"(gaddr) : "memory");
}
__device__ __forceinline__ void wait_async_le1() {
  asm volatile("s_wait_asynccnt 1" ::: "memory");
}
__device__ __forceinline__ void wait_async_le0() {
  asm volatile("s_wait_asynccnt 0" ::: "memory");
}

// ---------------------------------------------------------------------------
// Kernel A: per (b,g) row -> overlaps, align metric, zeroed pos mask, top-13
// ---------------------------------------------------------------------------
__global__ void __launch_bounds__(256)
row_metrics_topk(const float* __restrict__ pred_bboxes,
                 const float* __restrict__ pred_scores,
                 const float* __restrict__ priors,
                 const int*   __restrict__ gt_labels,
                 const float* __restrict__ gt_bboxes,
                 const float* __restrict__ pad_flag,
                 float* __restrict__ ovl,
                 float* __restrict__ met,
                 float* __restrict__ pos)
{
  __shared__ float s_box[2][TILE*4];   // double-buffered pred-box tiles
  __shared__ float s_v[256];
  __shared__ int   s_i[256];

  const int tid = threadIdx.x;
  const int row = blockIdx.x;          // b*NG + g
  const int b   = row >> 6;

  const float gx0 = gt_bboxes[row*4+0];
  const float gy0 = gt_bboxes[row*4+1];
  const float gx1 = gt_bboxes[row*4+2];
  const float gy1 = gt_bboxes[row*4+3];
  const int   lbl = gt_labels[row];
  const float pad = pad_flag[row];
  const float areag = fmaxf(gx1-gx0,0.f)*fmaxf(gy1-gy0,0.f);

  const size_t base = (size_t)row * NP;
  const float* pb = pred_bboxes + (size_t)b * NP * 4;
  const float* ps = pred_scores + (size_t)b * NP * NC + lbl;

  float lmet[NPT];                 // per-thread metric slice (slot j -> p = j*256+tid)
  unsigned long long ingmask = 0ull;

  // async-stage tile 0 into LDS
  {
    unsigned loff = (unsigned)(unsigned long long)&s_box[0][tid*4];
    async_copy_b128(loff, pb + (size_t)tid*4);
  }

  for (int t = 0; t < NPT; ++t) {
    const int buf = t & 1;
    if (t + 1 < NPT) {               // prefetch next tile (clamped: uniform count)
      int pn = (t+1)*TILE + tid; if (pn > NP-1) pn = NP-1;
      unsigned loff = (unsigned)(unsigned long long)&s_box[buf^1][tid*4];
      async_copy_b128(loff, pb + (size_t)pn*4);
      wait_async_le1();              // oldest (current tile) done; loads retire in order
    } else {
      wait_async_le0();
    }
    const int p = t*TILE + tid;
    if (p < NP) {
      const float bx0 = s_box[buf][tid*4+0];
      const float by0 = s_box[buf][tid*4+1];
      const float bx1 = s_box[buf][tid*4+2];
      const float by1 = s_box[buf][tid*4+3];
      const float iw = fminf(gx1,bx1) - fmaxf(gx0,bx0);
      const float ih = fminf(gy1,by1) - fmaxf(gy0,by0);
      const float inter = fmaxf(iw,0.f)*fmaxf(ih,0.f);
      const float areap = fmaxf(bx1-bx0,0.f)*fmaxf(by1-by0,0.f);
      const float iou   = inter / (areag + areap - inter + 1e-9f);
      const float sc    = ps[(size_t)p*NC];
      const float i2    = iou*iou;
      const float align = sc * (i2*i2*i2);           // score^1 * iou^6
      const float px = priors[p*4+0];
      const float py = priors[p*4+1];
      const float mind = fminf(fminf(px-gx0, py-gy0), fminf(gx1-px, gy1-py));
      const bool ing = mind > 1e-9f;
      if (ing) ingmask |= (1ull << t);
      ovl[base+p] = iou;
      met[base+p] = align;       // pure align metric (am = align * pos later)
      pos[base+p] = 0.f;
      lmet[t] = ing ? align : 0.f;
    } else {
      lmet[t] = -1e30f;
    }
  }

  // iterative block-argmax top-13 over the row, scatter pos = in_gts * pad
  for (int r = 0; r < KTOP; ++r) {
    float bv = -1e30f; int bj = 0;
    #pragma unroll
    for (int j = 0; j < NPT; ++j) if (lmet[j] > bv) { bv = lmet[j]; bj = j; }
    s_v[tid] = bv; s_i[tid] = bj*TILE + tid;
    __syncthreads();
    for (int s = 128; s > 0; s >>= 1) {
      if (tid < s && s_v[tid+s] > s_v[tid]) { s_v[tid] = s_v[tid+s]; s_i[tid] = s_i[tid+s]; }
      __syncthreads();
    }
    const int win = s_i[0];
    if ((win & (TILE-1)) == tid) {       // owner thread retires the winner
      const int wj = win >> 8;
      #pragma unroll
      for (int j = 0; j < NPT; ++j) if (j == wj) lmet[j] = -1e30f;
      const float ing = (float)((ingmask >> wj) & 1ull);
      pos[base + win] = ing * pad;
    }
    __syncthreads();
  }
}

// ---------------------------------------------------------------------------
// Kernel B1: WMMA fg = ones(1xG) x pos(GxP); resolve multi-assigned anchors;
//            gather labels/bboxes, emit fg flag + gt index.
// ---------------------------------------------------------------------------
__global__ void __launch_bounds__(256)
resolve_and_gather(float* __restrict__ pos,
                   const float* __restrict__ ovl,
                   const int*   __restrict__ gt_labels,
                   const float* __restrict__ gt_bboxes,
                   float* __restrict__ out_labels,
                   float* __restrict__ out_bboxes,
                   float* __restrict__ out_fg,
                   int*   __restrict__ gti)
{
  __shared__ float fgbuf[8][16];
  const int tid  = threadIdx.x;
  const int wave = tid >> 5;
  const int lane = tid & 31;
  const int grp  = blockIdx.x * 8 + wave;   // 8400 groups of 16 anchors
  const int a0   = grp * 16;
  const int b    = a0 / NP;
  const int p0   = a0 - b * NP;

  // A-matrix 16x32 f16 layout: lane m=lane&15 is row M; K halves split by lane>=16
  const int m  = lane & 15;
  const int kb = (lane >> 4) * 8;
  const size_t colb = (size_t)b * NG * NP + p0 + m;

  v16h ones;
  #pragma unroll
  for (int e = 0; e < 16; ++e) ones[e] = (_Float16)1.0f;

  v8f acc = {0.f,0.f,0.f,0.f,0.f,0.f,0.f,0.f};
  #pragma unroll
  for (int g0 = 0; g0 < NG; g0 += 32) {
    v16h a;
    #pragma unroll
    for (int e = 0; e < 16; ++e) {
      const int K = g0 + ((e < 8) ? (kb + e) : (16 + kb + (e - 8)));
      a[e] = (_Float16)pos[colb + (size_t)K * NP];
    }
    // D[m,n] = sum_k pos[m,k] * 1 + C  -> every column holds fg
    acc = __builtin_amdgcn_wmma_f32_16x16x32_f16(false, a, false, ones,
                                                 (short)0, acc, false, false);
  }
  // C/D layout: lane L, VGPR r -> M = r + 8*(L>=16), N = L&15. Take N==0 lanes.
  if ((lane & 15) == 0) {
    const int half = lane >> 4;
    #pragma unroll
    for (int r2 = 0; r2 < 8; ++r2) fgbuf[wave][half*8 + r2] = acc[r2];
  }
  __syncthreads();

  if (lane < 16) {
    const int p  = p0 + lane;
    const size_t col = (size_t)b * NG * NP + p;
    const float fg = fgbuf[wave][lane];
    int   gidx = 0;
    float fgo  = 0.f;
    if (fg > 1.5f) {                       // multi-assigned: keep overlap argmax
      float bo = -1.f; int gm = 0;
      for (int g = 0; g < NG; ++g) {
        const float o = ovl[col + (size_t)g*NP];
        if (o > bo) { bo = o; gm = g; }
      }
      for (int g = 0; g < NG; ++g) pos[col + (size_t)g*NP] = (g == gm) ? 1.f : 0.f;
      gidx = gm; fgo = 1.f;
    } else if (fg > 0.5f) {                // unique: first positive g
      for (int g = 0; g < NG; ++g) {
        if (pos[col + (size_t)g*NP] > 0.f) { gidx = g; break; }
      }
      fgo = 1.f;
    }
    const int gr = b*NG + gidx;
    const int o  = b*NP + p;
    const int lv = gt_labels[gr];
    out_labels[o] = (float)(lv > 0 ? lv : 0);
    out_bboxes[o*4+0] = gt_bboxes[gr*4+0];
    out_bboxes[o*4+1] = gt_bboxes[gr*4+1];
    out_bboxes[o*4+2] = gt_bboxes[gr*4+2];
    out_bboxes[o*4+3] = gt_bboxes[gr*4+3];
    out_fg[o] = fgo;
    gti[o] = gidx;
  }
}

// ---------------------------------------------------------------------------
// Kernel B2: per (b,g): pos_align = max_p align*pos, pos_ovl = max_p iou*pos
// ---------------------------------------------------------------------------
__global__ void __launch_bounds__(256)
row_pos_max(const float* __restrict__ met,
            const float* __restrict__ ovl,
            const float* __restrict__ pos,
            float* __restrict__ palign,
            float* __restrict__ povl)
{
  __shared__ float sa[256], so[256];
  const int tid = threadIdx.x;
  const int row = blockIdx.x;
  const size_t base = (size_t)row * NP;
  float ma = 0.f, mo = 0.f;
  for (int p = tid; p < NP; p += 256) {
    const float pm = pos[base+p];
    ma = fmaxf(ma, met[base+p] * pm);
    mo = fmaxf(mo, ovl[base+p] * pm);
  }
  sa[tid] = ma; so[tid] = mo; __syncthreads();
  for (int s = 128; s > 0; s >>= 1) {
    if (tid < s) { sa[tid] = fmaxf(sa[tid], sa[tid+s]); so[tid] = fmaxf(so[tid], so[tid+s]); }
    __syncthreads();
  }
  if (tid == 0) { palign[row] = sa[0]; povl[row] = so[0]; }
}

// ---------------------------------------------------------------------------
// Kernel B3: norm + one-hot scores. Final pos column has <=1 nonzero, so the
// max over g collapses to the gt_idx term.
// ---------------------------------------------------------------------------
__global__ void __launch_bounds__(256)
write_scores(const float* __restrict__ met,
             const float* __restrict__ palign,
             const float* __restrict__ povl,
             const int*   __restrict__ gti,
             const float* __restrict__ out_fg,
             const float* __restrict__ out_labels,
             float* __restrict__ out_scores)
{
  const int idx = blockIdx.x*256 + threadIdx.x;
  if (idx >= BP) return;
  const int b = idx / NP;
  const int p = idx - b*NP;
  const float fg = out_fg[idx];
  float norm = 0.f; int lbl = -1;
  if (fg > 0.f) {
    const int gr = b*NG + gti[idx];
    norm = met[(size_t)gr*NP + p] * povl[gr] / (palign[gr] + 1e-7f);
    lbl  = (int)out_labels[idx];
  }
  float* o = out_scores + (size_t)idx * NC;
  #pragma unroll 4
  for (int c = 0; c < NC; ++c) o[c] = (c == lbl) ? norm : 0.f;
}

// ---------------------------------------------------------------------------
extern "C" void kernel_launch(void* const* d_in, const int* in_sizes, int n_in,
                              void* d_out, int out_size, void* d_ws, size_t ws_size,
                              hipStream_t stream)
{
  const float* pred_bboxes = (const float*)d_in[0];   // (B,P,4)
  const float* pred_scores = (const float*)d_in[1];   // (B,P,C)
  const float* priors      = (const float*)d_in[2];   // (P,4)
  const int*   gt_labels   = (const int*)  d_in[3];   // (B,G,1)
  const float* gt_bboxes   = (const float*)d_in[4];   // (B,G,4)
  const float* pad_flag    = (const float*)d_in[5];   // (B,G,1)

  float* ws  = (float*)d_ws;
  float* ovl = ws;                         // (B,G,P)
  float* met = ws + BGP;                   // (B,G,P)
  float* pos = ws + 2*BGP;                 // (B,G,P)
  float* pal = ws + 3*BGP;                 // (B,G)
  float* pov = pal + NB*NG;                // (B,G)
  int*   gti = (int*)(pov + NB*NG);        // (B,P)

  float* out        = (float*)d_out;       // tuple concatenated flat
  float* out_labels = out;                 // (B,P)
  float* out_bboxes = out + BP;            // (B,P,4)
  float* out_scores = out + 5*(size_t)BP;  // (B,P,C)
  float* out_fg     = out + 85*(size_t)BP; // (B,P)

  row_metrics_topk<<<NB*NG, 256, 0, stream>>>(pred_bboxes, pred_scores, priors,
                                              gt_labels, gt_bboxes, pad_flag,
                                              ovl, met, pos);
  resolve_and_gather<<<(NB*NP)/(16*8), 256, 0, stream>>>(pos, ovl, gt_labels, gt_bboxes,
                                                         out_labels, out_bboxes, out_fg, gti);
  row_pos_max<<<NB*NG, 256, 0, stream>>>(met, ovl, pos, pal, pov);
  write_scores<<<(BP+255)/256, 256, 0, stream>>>(met, pal, pov, gti, out_fg,
                                                 out_labels, out_scores);
}
